// CapsNet_1743756722633
// MI455X (gfx1250) — matrix operand order repaired and checked by reading, not compiled, not run on hardware
//
#include <hip/hip_runtime.h>
#include <hip/hip_bf16.h>
#include <math.h>

// ---------------------------------------------------------------------------
// CapsNet forward on gfx1250 (MI455X), compile-only build.
//  K0 : f32->bf16 pre-conversion of input / conv1_w / prim_w (one-time, small)
//  K1 : conv1 (implicit GEMM, M=352352 N=256 K=4800) bf16 WMMA, fused relu.
//       Epilogue scatters straight into pre-im2col layout A2[m2][k] for the
//       second conv (legal because prim conv stride==kernel, patches disjoint)
//       -> y1 is 3456 x 25600 bf16 = 177MB, ~L2 resident.
//  K2 : primary-caps conv (M=3456 N=256 K=25600) bf16 WMMA, split-K x8,
//       partials combined with global_atomic_add_f32 into bias-initialized s2.
//  K3 : reshape + squash -> u (B,3456,8) f32
//  K4 : per-batch routing (b-logits LDS-resident, ds_add_f32 reductions)
// Tile staging: double-buffered LDS fed by GLOBAL_LOAD_ASYNC_TO_LDS_B64/B128
// (ASYNCcnt), next tile issued before s_wait_asynccnt N on the current one.
// ---------------------------------------------------------------------------

typedef __attribute__((ext_vector_type(16))) __bf16 v16bf;
typedef __attribute__((ext_vector_type(8)))  float  v8f;

union Frag16 { v16bf v; unsigned u[8]; };

// f32 -> bf16 round-to-nearest-even
__device__ __forceinline__ unsigned short f2bf(float f) {
    unsigned x = __builtin_bit_cast(unsigned, f);
    unsigned r = x + 0x7fffu + ((x >> 16) & 1u);
    return (unsigned short)(r >> 16);
}

// async copies: global (saddr + voff bytes) -> LDS[ldsaddr]
__device__ __forceinline__ void async_b64(unsigned ldsaddr, unsigned voff,
                                          const void* base) {
    asm volatile("global_load_async_to_lds_b64 %0, %1, %2"
                 :: "v"(ldsaddr), "v"(voff), "s"(base) : "memory");
}
__device__ __forceinline__ void async_b128(unsigned ldsaddr, unsigned voff,
                                           const void* base) {
    asm volatile("global_load_async_to_lds_b128 %0, %1, %2"
                 :: "v"(ldsaddr), "v"(voff), "s"(base) : "memory");
}
template <int N>
__device__ __forceinline__ void wait_async() {
    asm volatile("s_wait_asynccnt %0" :: "i"(N) : "memory");
}

#define BM 128
#define BN 128
#define BK 32

// ---- K0: elementwise f32 -> bf16 ----
__global__ __launch_bounds__(256) void f32_to_bf16(
    const float* __restrict__ s, unsigned short* __restrict__ d, int n)
{
    const int i = blockIdx.x * 256 + threadIdx.x;
    if (i < n) d[i] = f2bf(s[i]);
}

// ---- init s2 with the primary-caps bias (split-K partials atomically add) --
__global__ __launch_bounds__(256) void init_s2(
    float* __restrict__ s2, const float* __restrict__ prim_b)
{
    const int idx = blockIdx.x * 256 + threadIdx.x;   // (B,256,9,12) flat
    if (idx < 32 * 256 * 108) s2[idx] = prim_b[(idx / 108) & 255];
}

// MODE 0: conv1   : A(m,k)=in_bf[b,oh+kh,ow*4+kw], B(k,n)=w1_bf[n,k]
// MODE 1: primconv: A(m,k)=y1[m*25600+k] (pre-im2col), B(k,n)=w2_bf[n,k]
template <int MODE>
__global__ __launch_bounds__(256) void conv_gemm(
    const unsigned short* __restrict__ srcA,    // bf16 activations
    const unsigned short* __restrict__ wgt,     // bf16 weights (n-major, K contig)
    const float* __restrict__ bias,             // MODE0 only
    unsigned short* __restrict__ dst_bf,        // MODE0: y1 (pre-im2col layout)
    float* __restrict__ dst_f32)                // MODE1: s2 (atomic accumulate)
{
    constexpr int Mtot = (MODE == 0) ? 352352 : 3456;
    constexpr int Ktot = (MODE == 0) ? 4800   : 25600;
    constexpr int KCH  = 3200;                       // split-K chunk (MODE1)
    constexpr int NST  = (MODE == 0) ? 6 : 4;        // async ops per thread/stage

    __shared__ unsigned As[2][BM][BK / 2];           // double-buffered bf16 pairs
    __shared__ unsigned Bs[2][BN][BK / 2];

    const int tid  = threadIdx.x;
    const int lane = tid & 31;
    const int wid  = tid >> 5;
    const int wm   = wid & 3;                        // 4 waves x 32 rows
    const int wn   = wid >> 2;                       // 2 waves x 64 cols
    const int m0   = blockIdx.x * BM;
    const int n0   = blockIdx.y * BN;
    const int kbeg = (MODE == 1) ? (int)blockIdx.z * KCH : 0;
    const int kend = (MODE == 1) ? kbeg + KCH : Ktot;

    const int half = (lane < 16) ? 0 : 1;
    const int l15  = lane & 15;

    const unsigned asb = (unsigned)(size_t)&As[0][0][0];  // LDS byte address
    const unsigned bsb = (unsigned)(size_t)&Bs[0][0][0];

    // stage one (A,B) tile pair for K-slice kk into LDS buffer `buf`
    auto stage = [&](int buf, int kk) {
        const unsigned ab = asb + (unsigned)buf * (BM * 64);
        const unsigned bb = bsb + (unsigned)buf * (BN * 64);
        if (MODE == 0) {
            // A: 1024 x b64 chunks (4 bf16 each); kw stays in-row (160%32==0)
            #pragma unroll
            for (int it = 0; it < 4; ++it) {
                const int t = it * 256 + tid;
                const int row = t >> 3, cc = t & 7;
                int m = m0 + row;
                m = (m < Mtot) ? m : (Mtot - 1);     // clamp: row never stored
                const int k = kk + cc * 4;
                const int ow = m % 121, rem = m / 121;
                const int oh = rem % 91, b = rem / 91;
                const int kh = k / 160, kw2 = k - kh * 160;
                const size_t idx =
                    ((size_t)b * 120 + oh + kh) * 640 + ow * 4 + kw2;
                async_b64(ab + 8u * (unsigned)t, (unsigned)(idx * 2), srcA);
            }
        } else {
            // A: 512 x b128 chunks; pre-im2col layout -> offset is linear in k
            #pragma unroll
            for (int it = 0; it < 2; ++it) {
                const int t = it * 256 + tid;
                const int row = t >> 2, cc = t & 3;
                int m = m0 + row;
                m = (m < Mtot) ? m : (Mtot - 1);
                const unsigned voff =
                    (unsigned)(((size_t)m * 25600 + kk + cc * 8) * 2);
                async_b128(ab + 16u * (unsigned)t, voff, srcA);
            }
        }
        // B: 512 x b128 chunks (weights K-contiguous, 16B aligned)
        #pragma unroll
        for (int it = 0; it < 2; ++it) {
            const int t = it * 256 + tid;
            const int col = t >> 2, cc = t & 3;
            const unsigned voff =
                (unsigned)(((size_t)(n0 + col) * Ktot + kk + cc * 8) * 2);
            async_b128(bb + 16u * (unsigned)t, voff, wgt);
        }
    };

    v8f acc[2][4];
    for (int i = 0; i < 2; ++i)
        for (int j = 0; j < 4; ++j)
            for (int e = 0; e < 8; ++e) acc[i][j][e] = 0.f;

    stage(0, kbeg);                                  // prologue fill
    int buf = 0;
    for (int kk = kbeg; kk < kend; kk += BK, buf ^= 1) {
        const bool more = (kk + BK) < kend;
        if (more) {
            stage(buf ^ 1, kk + BK);                 // overlap next copy w/ compute
            __builtin_prefetch(                      // and hint the K-slice after
                (const void*)(wgt + (size_t)(n0 + (tid & 127)) * Ktot + kk + BK),
                0, 2);
            wait_async<NST>();                       // cur stage landed (in-order)
        } else {
            wait_async<0>();
        }
        __syncthreads();                             // whole tile visible

        // ---- A(16x32)/B(32x16) fragments: contiguous k = 16*half + s, same for A and B ----
        Frag16 af[2], bfr[4];
        for (int mi = 0; mi < 2; ++mi) {
            const int arow = wm * 32 + mi * 16 + l15;
            #pragma unroll
            for (int p = 0; p < 8; ++p) af[mi].u[p] = As[buf][arow][half * 8 + p];
        }
        for (int ni = 0; ni < 4; ++ni) {
            const int bcol = wn * 64 + ni * 16 + l15;
            #pragma unroll
            for (int p = 0; p < 8; ++p) bfr[ni].u[p] = Bs[buf][bcol][half * 8 + p];
        }
        #pragma unroll
        for (int mi = 0; mi < 2; ++mi)
            #pragma unroll
            for (int ni = 0; ni < 4; ++ni)
                acc[mi][ni] = __builtin_amdgcn_wmma_f32_16x16x32_bf16(
                    false, af[mi].v, false, bfr[ni].v,
                    (short)0, acc[mi][ni], false, false);
        __syncthreads();       // all waves done reading before buf is re-staged
    }

    // ---- epilogue: C frag -> lane = col, VGPR r = row (+8 for hi half) ----
    for (int mi = 0; mi < 2; ++mi) {
        for (int ni = 0; ni < 4; ++ni) {
            const int col = n0 + wn * 64 + ni * 16 + l15;
            const float bv = (MODE == 0) ? bias[col] : 0.f;
            #pragma unroll
            for (int r = 0; r < 8; ++r) {
                const int m = m0 + wm * 32 + mi * 16 + r + half * 8;
                if (m < Mtot) {
                    if (MODE == 0) {
                        float v = acc[mi][ni][r] + bv;
                        v = v > 0.f ? v : 0.f;                 // fused relu
                        const int ow = m % 121, rem = m / 121;
                        const int oh = rem % 91, b = rem / 91;
                        // scatter into pre-im2col A2[m2][c*100+r*10+s];
                        // 10x10 patches are disjoint (stride==kernel)
                        const int oh2 = oh / 10, rr = oh - oh2 * 10;
                        const int ow2 = ow / 10, ssp = ow - ow2 * 10;
                        if (oh2 < 9 && ow2 < 12) {
                            const int m2 = b * 108 + oh2 * 12 + ow2;
                            dst_bf[(size_t)m2 * 25600 + col * 100 + rr * 10 + ssp]
                                = f2bf(v);
                        }
                    } else {                                   // split-K partial
                        const int ow = m % 12, rem = m / 12;
                        const int oh = rem % 9, b = rem / 9;
                        atomicAdd(&dst_f32[(((size_t)b * 256 + col) * 9 + oh) * 12
                                           + ow],
                                  acc[mi][ni][r]);   // global_atomic_add_f32
                    }
                }
            }
        }
    }
}

// ---- K3: (B,256,9,12) -> reshape/transpose -> squash -> u (B,3456,8) ----
__global__ __launch_bounds__(256) void squash_primary(
    const float* __restrict__ s2, float* __restrict__ u)
{
    const int idx = blockIdx.x * 256 + threadIdx.x;   // over 32*3456
    if (idx >= 32 * 3456) return;
    const int b = idx / 3456, i = idx - b * 3456;
    const int g = i / 108, rem = i - g * 108;
    const int oh = rem / 12, ow = rem - oh * 12;
    float x[8], l2 = 0.f;
    #pragma unroll
    for (int d = 0; d < 8; ++d) {
        x[d] = s2[(((size_t)b * 256 + g * 8 + d) * 9 + oh) * 12 + ow];
        l2 += x[d] * x[d];
    }
    const float sc = l2 / ((1.f + l2) * sqrtf(l2));
    #pragma unroll
    for (int d = 0; d < 8; ++d) u[(size_t)idx * 8 + d] = x[d] * sc;
}

// ---- K4: dynamic routing + heads, one workgroup per batch element ----
__global__ __launch_bounds__(256) void routing_kernel(
    const float* __restrict__ u,        // (B,3456,8)
    const float* __restrict__ caps_w,   // (3456,8,80)
    const float* __restrict__ b_route,  // (3456,5)
    const float* __restrict__ pred_w,   // (25,16)
    const float* __restrict__ pred_b,   // (25)
    const float* __restrict__ eos_w,    // (1,16)
    const float* __restrict__ eos_b,    // (1)
    float* __restrict__ out)            // (B,5,26)
{
    __shared__ float bb[3456 * 5];      // 69 KB routing logits, LDS-resident
    __shared__ float vv[80];
    __shared__ float ss[80];

    const int b = blockIdx.x;
    const int tid = threadIdx.x;

    for (int t = tid; t < 3456 * 5; t += 256) bb[t] = b_route[t];
    if (tid < 80) { ss[tid] = 0.f; vv[tid] = 0.f; }
    __syncthreads();

    for (int pass = 0; pass < 4; ++pass) {   // pass0 = initial, 1..3 = iters
        for (int i = tid; i < 3456; i += 256) {
            float ui[8];
            const float* pu = u + ((size_t)b * 3456 + i) * 8;
            #pragma unroll
            for (int c8 = 0; c8 < 8; ++c8) ui[c8] = pu[c8];

            float up[80];                    // u_predict[i] recomputed on the fly
            const float* cw = caps_w + (size_t)i * 640;
            #pragma unroll
            for (int j = 0; j < 80; ++j) {
                float s = 0.f;
                #pragma unroll
                for (int c8 = 0; c8 < 8; ++c8) s += ui[c8] * cw[c8 * 80 + j];
                up[j] = s;
            }

            float* bbi = &bb[i * 5];
            if (pass > 0) {
                #pragma unroll
                for (int o = 0; o < 5; ++o) {
                    float dv = 0.f;
                    #pragma unroll
                    for (int d = 0; d < 16; ++d) dv += up[o * 16 + d] * vv[o * 16 + d];
                    bbi[o] += dv;
                }
            }
            float mx = bbi[0];
            #pragma unroll
            for (int o = 1; o < 5; ++o) mx = fmaxf(mx, bbi[o]);
            float e[5], sum = 0.f;
            #pragma unroll
            for (int o = 0; o < 5; ++o) { e[o] = __expf(bbi[o] - mx); sum += e[o]; }
            const float inv = 1.f / sum;
            #pragma unroll
            for (int o = 0; o < 5; ++o) {
                const float c = e[o] * inv;
                #pragma unroll
                for (int d = 0; d < 16; ++d)
                    atomicAdd(&ss[o * 16 + d], c * up[o * 16 + d]);  // ds_add_f32
            }
        }
        __syncthreads();
        if (tid < 5) {                       // squash s -> v
            float l2 = 0.f;
            #pragma unroll
            for (int d = 0; d < 16; ++d) l2 += ss[tid * 16 + d] * ss[tid * 16 + d];
            const float sc = l2 / ((1.f + l2) * sqrtf(l2));
            #pragma unroll
            for (int d = 0; d < 16; ++d) vv[tid * 16 + d] = ss[tid * 16 + d] * sc;
        }
        __syncthreads();
        if (tid < 80) ss[tid] = 0.f;
        __syncthreads();
    }

    if (tid < 5) {                           // heads + log_softmax (26 logits)
        float lg[26];
        #pragma unroll
        for (int j = 0; j < 25; ++j) {
            float s = pred_b[j];
            #pragma unroll
            for (int d = 0; d < 16; ++d) s += vv[tid * 16 + d] * pred_w[j * 16 + d];
            lg[j] = s;
        }
        {
            float s = eos_b[0];
            #pragma unroll
            for (int d = 0; d < 16; ++d) s += vv[tid * 16 + d] * eos_w[d];
            lg[25] = s;
        }
        float mx = lg[0];
        #pragma unroll
        for (int j = 1; j < 26; ++j) mx = fmaxf(mx, lg[j]);
        float lse = 0.f;
        #pragma unroll
        for (int j = 0; j < 26; ++j) lse += __expf(lg[j] - mx);
        lse = __logf(lse) + mx;
        #pragma unroll
        for (int j = 0; j < 26; ++j)
            out[((size_t)b * 5 + tid) * 26 + j] = lg[j] - lse;
    }
}

extern "C" void kernel_launch(void* const* d_in, const int* in_sizes, int n_in,
                              void* d_out, int out_size, void* d_ws, size_t ws_size,
                              hipStream_t stream) {
    const float* input   = (const float*)d_in[0];
    const float* conv1_w = (const float*)d_in[1];
    const float* conv1_b = (const float*)d_in[2];
    const float* prim_w  = (const float*)d_in[3];
    const float* prim_b  = (const float*)d_in[4];
    const float* caps_w  = (const float*)d_in[5];
    const float* b_route = (const float*)d_in[6];
    const float* pred_w  = (const float*)d_in[7];
    const float* pred_b  = (const float*)d_in[8];
    const float* eos_w   = (const float*)d_in[9];
    const float* eos_b   = (const float*)d_in[10];
    (void)in_sizes; (void)n_in; (void)out_size; (void)ws_size;

    // ---- workspace carve-up (256B-aligned sections) ----
    const size_t ALIGN = 255;
    char* p = (char*)d_ws;
    auto carve = [&](size_t bytes) {
        char* r = p; p += (bytes + ALIGN) & ~ALIGN; return r;
    };
    unsigned short* y1    = (unsigned short*)carve(3456ull * 25600 * 2); // 177MB
    float*          s2    = (float*)carve(32ull * 256 * 108 * 4);
    float*          u     = (float*)carve(32ull * 3456 * 8 * 4);
    unsigned short* in_bf = (unsigned short*)carve(2457600ull * 2);
    unsigned short* w1_bf = (unsigned short*)carve(1228800ull * 2);
    unsigned short* w2_bf = (unsigned short*)carve(6553600ull * 2);

    // K0: bf16 pre-conversion (one-time, tiny vs. the 177MB y1 stream)
    f32_to_bf16<<<(2457600 + 255) / 256, 256, 0, stream>>>(input,   in_bf, 2457600);
    f32_to_bf16<<<(1228800 + 255) / 256, 256, 0, stream>>>(conv1_w, w1_bf, 1228800);
    f32_to_bf16<<<(6553600 + 255) / 256, 256, 0, stream>>>(prim_w,  w2_bf, 6553600);
    // s2 must be bias-initialized (split-K partials atomically accumulate)
    init_s2<<<(32 * 256 * 108 + 255) / 256, 256, 0, stream>>>(s2, prim_b);

    // K1: conv1 + relu -> pre-im2col y1  (2753 M-tiles x 2 N-tiles)
    dim3 g1(2753, 2, 1);
    conv_gemm<0><<<g1, 256, 0, stream>>>(in_bf, w1_bf, conv1_b, y1, nullptr);
    // K2: primary caps conv, split-K x8 -> 432 workgroups
    dim3 g2(27, 2, 8);
    conv_gemm<1><<<g2, 256, 0, stream>>>(y1, w2_bf, nullptr, nullptr, s2);
    // K3: squash
    squash_primary<<<(32 * 3456 + 255) / 256, 256, 0, stream>>>(s2, u);
    // K4: routing + heads
    routing_kernel<<<32, 256, 0, stream>>>(u, caps_w, b_route, pred_w, pred_b,
                                           eos_w, eos_b, (float*)d_out);
}
